// _LTransformerDecoderBlock_53687091200455
// MI455X (gfx1250) — matrix operand order repaired and checked by reading, not compiled, not run on hardware
//
#include <hip/hip_runtime.h>
#include <cstdint>
#include <cstddef>

typedef __attribute__((ext_vector_type(16))) _Float16 v16h;
typedef __attribute__((ext_vector_type(8)))  _Float16 v8h;
typedef __attribute__((ext_vector_type(8)))  float    v8f;

#define D_MODEL 1024
#define NHEAD   16
#define HD      64
#define BB      2
#define SS      2048
#define MROWS   (BB*SS)          /* 4096 rows of (b,s) */
#define INV_SCALE (1.0f/64.0f)   /* SCALE = 2*sqrt(1024) = 64 */

// ---------------------------------------------------------------------------
// CDNA5 async global->LDS copy (ASYNCcnt-tracked, ISA 08_async_tensor §4)
// ---------------------------------------------------------------------------
__device__ __forceinline__ void g2l_async_b128(void* ldsdst, const void* gsrc) {
    unsigned l = (unsigned)(uintptr_t)ldsdst;              // LDS byte address
    unsigned long long g = (unsigned long long)(uintptr_t)gsrc;
    asm volatile("global_load_async_to_lds_b128 %0, %1, off"
                 :: "v"(l), "v"(g) : "memory");
}
__device__ __forceinline__ void wait_async0() {
    asm volatile("s_wait_asynccnt 0x0" ::: "memory");
}

// ---------------------------------------------------------------------------
// Fragment helpers (CDNA5 WMMA 16x16x32 f16, wave32; layouts per ISA 7.12.2)
// ---------------------------------------------------------------------------
__device__ __forceinline__ v16h frag16(const _Float16* p0, const _Float16* p1) {
    v8h a = *(const v8h*)p0;
    v8h b = *(const v8h*)p1;
    return __builtin_shufflevector(a, b, 0,1,2,3,4,5,6,7,8,9,10,11,12,13,14,15);
}
// A fragment: row ptr for this lane's M (=lane%16); covers K = k0..k0+31.
__device__ __forceinline__ v16h loadA(const _Float16* row, int k0, int hf) {
    return frag16(row + k0 + hf*8, row + k0 + 16 + hf*8);
}
// B fragment from K-contiguous layout: row ptr for this lane's N (=lane%16);
// lanes 0-15 hold K=k0..k0+7,k0+16..k0+23; lanes 16-31 hold K=k0+8..k0+15,k0+24..k0+31
__device__ __forceinline__ v16h loadB(const _Float16* rowN, int k0, int hf) {
    const _Float16* p = rowN + k0 + hf*8;
    return frag16(p, p + 16);
}
__device__ __forceinline__ v8f wmma_f16(v16h a, v16h b, v8f c) {
    return __builtin_amdgcn_wmma_f32_16x16x32_f16(false, a, false, b, (short)0, c, false, false);
}

__device__ __forceinline__ float block_sum(float v, float* red) {
    #pragma unroll
    for (int off = 16; off; off >>= 1) v += __shfl_xor(v, off, 32);
    if ((threadIdx.x & 31) == 0) red[threadIdx.x >> 5] = v;
    __syncthreads();
    float r = (threadIdx.x < 8) ? red[threadIdx.x] : 0.0f;
    #pragma unroll
    for (int off = 4; off; off >>= 1) r += __shfl_xor(r, off, 32);
    if (threadIdx.x == 0) red[0] = r;
    __syncthreads();
    float out = red[0];
    __syncthreads();
    return out;
}

// ---------------------------------------------------------------------------
// Weight pack: dst[n][k] = (k<srcR && n<srcC) ? src[k][n] : 0   (f32 -> f16 T)
// ---------------------------------------------------------------------------
__global__ void pack_wT(const float* __restrict__ src, _Float16* __restrict__ dst,
                        int srcR, int srcC, int dstN, int dstK)
{
    size_t total = (size_t)dstN * (size_t)dstK;
    for (size_t idx = (size_t)blockIdx.x * blockDim.x + threadIdx.x; idx < total;
         idx += (size_t)gridDim.x * blockDim.x) {
        int n = (int)(idx / dstK), k = (int)(idx % dstK);
        float v = (k < srcR && n < srcC) ? src[(size_t)k * srcC + n] : 0.0f;
        dst[idx] = (_Float16)v;
    }
}

// ---------------------------------------------------------------------------
// lrmsnorm over x[...,1:] (1023 dims) -> f16 padded row of 1024
// ---------------------------------------------------------------------------
__global__ __launch_bounds__(256)
void lrmsnorm_pack(const float* __restrict__ x, const float* __restrict__ wgt,
                   _Float16* __restrict__ out)
{
    __shared__ float red[8];
    int row = blockIdx.x;
    const float* xr = x + (size_t)row * D_MODEL;
    float ss = 0.0f;
    for (int j = threadIdx.x; j < D_MODEL - 1; j += 256) { float s = xr[1 + j]; ss += s * s; }
    ss = block_sum(ss, red);
    float rinv = rsqrtf(ss / (float)(D_MODEL - 1) + 1e-6f);
    _Float16* orow = out + (size_t)row * D_MODEL;
    for (int j = threadIdx.x; j < D_MODEL; j += 256)
        orow[j] = (j < D_MODEL - 1) ? (_Float16)(xr[1 + j] * rinv * wgt[j]) : (_Float16)0.0f;
}

// ---------------------------------------------------------------------------
// Generic WMMA GEMM: C[M,N] = A[M,K] * BT[N,K]^T ; f16 in, f32 acc.
// Double-buffered LDS tiles filled by async global->LDS copies.
// mode 0: f32 store | 1: f16 store | 2: f16(gelu(v+bias)) | 3: f32(v+bias)
// ---------------------------------------------------------------------------
__global__ __launch_bounds__(256)
void gemm_wmma(const _Float16* __restrict__ A, const _Float16* __restrict__ BT,
               void* __restrict__ Cout, const float* __restrict__ bias,
               int M, int N, int Kd, int biasN, int mode)
{
    __shared__ __align__(16) _Float16 As[2][128][40];
    __shared__ __align__(16) _Float16 Bs[2][128][40];
    const int tid  = threadIdx.x;
    const int lane = tid & 31;
    const int w    = tid >> 5;
    const int wm   = (w & 1) * 64;
    const int wn   = (w >> 1) * 32;
    const int mbase = blockIdx.y * 128;
    const int nbase = blockIdx.x * 128;
    const int hf = lane >> 4, nl = lane & 15;

    v8f acc[4][2];
    #pragma unroll
    for (int i = 0; i < 4; i++)
        #pragma unroll
        for (int j = 0; j < 2; j++) acc[i][j] = (v8f){};

    auto stage = [&](int buf, int k0) {
        #pragma unroll
        for (int i = 0; i < 2; i++) {
            int c = tid + i * 256;           // 512 x 16B chunks per tile pair
            int m = c >> 2, ch = c & 3;
            g2l_async_b128(&As[buf][m][ch * 8],
                           A  + (size_t)(mbase + m) * Kd + k0 + ch * 8);
            g2l_async_b128(&Bs[buf][m][ch * 8],
                           BT + (size_t)(nbase + m) * Kd + k0 + ch * 8);
        }
    };

    stage(0, 0);
    wait_async0();
    __syncthreads();

    int cur = 0;
    for (int k0 = 0; k0 < Kd; k0 += 32) {
        if (k0 + 32 < Kd) stage(cur ^ 1, k0 + 32);   // overlap next slice

        v16h bf[2];
        #pragma unroll
        for (int j = 0; j < 2; j++) bf[j] = loadB(&Bs[cur][wn + j * 16 + nl][0], 0, hf);
        #pragma unroll
        for (int i = 0; i < 4; i++) {
            v16h af = loadA(&As[cur][wm + i * 16 + nl][0], 0, hf);
            #pragma unroll
            for (int j = 0; j < 2; j++) acc[i][j] = wmma_f16(af, bf[j], acc[i][j]);
        }
        wait_async0();
        __syncthreads();
        cur ^= 1;
    }

    #pragma unroll
    for (int i = 0; i < 4; i++)
        #pragma unroll
        for (int j = 0; j < 2; j++)
            #pragma unroll
            for (int r = 0; r < 8; r++) {
                int m = mbase + wm + i * 16 + r + 8 * hf;
                int n = nbase + wn + j * 16 + nl;
                float v = acc[i][j][r];
                if (mode == 0) {
                    ((float*)Cout)[(size_t)m * N + n] = v;
                } else if (mode == 1) {
                    ((_Float16*)Cout)[(size_t)m * N + n] = (_Float16)v;
                } else if (mode == 2) {
                    float b = (n < biasN) ? bias[n] : 0.0f;
                    float u = v + b;
                    float g = 0.5f * u * (1.0f + tanhf(0.7978845608f * (u + 0.044715f * u * u * u)));
                    ((_Float16*)Cout)[(size_t)m * N + n] = (_Float16)g;
                } else {
                    float b = (n < biasN) ? bias[n] : 0.0f;
                    ((float*)Cout)[(size_t)m * N + n] = v + b;
                }
            }
}

// ---------------------------------------------------------------------------
// Per-head Lorentz time coordinate: t = sqrt(sum_d v^2 + 1)
// ---------------------------------------------------------------------------
__global__ void lift_t(const _Float16* __restrict__ X, float* __restrict__ T)
{
    int idx = blockIdx.x * blockDim.x + threadIdx.x;
    if (idx >= MROWS * NHEAD) return;
    int row = idx >> 4, hh = idx & 15;
    const _Float16* p = X + (size_t)row * D_MODEL + hh * HD;
    float ss = 0.0f;
    #pragma unroll
    for (int c = 0; c < 8; c++) {
        v8h v = *(const v8h*)(p + c * 8);
        #pragma unroll
        for (int j = 0; j < 8; j++) { float f = (float)v[j]; ss += f * f; }
    }
    T[idx] = sqrtf(ss + 1.0f);
}

// ---------------------------------------------------------------------------
// Flash attention: per (128-query block, head, batch). 8 waves x 16 queries.
// score = (2 + 2*(q.k - tq*tk))/8 ; causal online softmax ; PV via WMMA;
// Lorentz proj() fused in epilogue; stores head-space (time dropped) as f16.
// ---------------------------------------------------------------------------
__global__ __launch_bounds__(256)
void attn_wmma(const _Float16* __restrict__ Qb, const _Float16* __restrict__ Kb,
               const _Float16* __restrict__ Vb,
               const float* __restrict__ TQ, const float* __restrict__ TK,
               const float* __restrict__ TV, _Float16* __restrict__ Oout)
{
    __shared__ __align__(16) _Float16 Ks[32][72];      // [kpos][d]
    __shared__ __align__(16) _Float16 Vt[64][40];      // [d][kpos]
    __shared__ __align__(16) _Float16 Pst[8][16][40];  // per-wave P staging
    __shared__ float tqs[128], tks[32], tvs[32];
    __shared__ float mrow[8][16], lrow[8][16], trow[8][16];

    const int tid = threadIdx.x, lane = tid & 31, w = tid >> 5;
    const int hf = lane >> 4, nl = lane & 15;
    const int qbase = blockIdx.x * 128;
    const int head  = blockIdx.y;
    const size_t rowbase = (size_t)blockIdx.z * SS;

    // Q fragments for this wave's 16 query rows (d 0..31, 32..63)
    const _Float16* qp = Qb + (rowbase + qbase + w * 16 + nl) * D_MODEL + head * HD;
    v16h qa0 = loadA(qp, 0, hf);
    v16h qa1 = loadA(qp, 32, hf);

    if (tid < 128) {
        tqs[tid] = TQ[(rowbase + qbase + tid) * NHEAD + head];
        mrow[tid >> 4][tid & 15] = -3.0e38f;
        lrow[tid >> 4][tid & 15] = 0.0f;
        trow[tid >> 4][tid & 15] = 0.0f;
    }
    v8f oac[4];
    #pragma unroll
    for (int dt = 0; dt < 4; dt++) oac[dt] = (v8f){};

    const int nkb = (qbase + 128) >> 5;   // causal: keys 0 .. qbase+127
    for (int kb = 0; kb < nkb; kb++) {
        int kpos0 = kb << 5;
        __syncthreads();
        {   // stage K via async copies; V via registers (transpose to [d][kpos])
            int kp = tid >> 3, ch = tid & 7;
            v8h vv = *(const v8h*)(Vb + (rowbase + kpos0 + kp) * D_MODEL + head * HD + ch * 8);
            g2l_async_b128(&Ks[kp][ch * 8],
                           Kb + (rowbase + kpos0 + kp) * D_MODEL + head * HD + ch * 8);
            #pragma unroll
            for (int j = 0; j < 8; j++) Vt[ch * 8 + j][kp] = vv[j];
        }
        if (tid < 32) {
            tks[tid] = TK[(rowbase + kpos0 + tid) * NHEAD + head];
            tvs[tid] = TV[(rowbase + kpos0 + tid) * NHEAD + head];
        }
        wait_async0();
        __syncthreads();

        // S = Q K^T for keys [kpos0, kpos0+32)
        v8f c0 = (v8f){}, c1 = (v8f){};
        c0 = wmma_f16(qa0, loadB(&Ks[nl][0],      0, hf), c0);
        c0 = wmma_f16(qa1, loadB(&Ks[nl][0],     32, hf), c0);
        c1 = wmma_f16(qa0, loadB(&Ks[16 + nl][0], 0, hf), c1);
        c1 = wmma_f16(qa1, loadB(&Ks[16 + nl][0],32, hf), c1);

        #pragma unroll
        for (int r = 0; r < 8; r++) {
            int m = r + 8 * hf;
            int qg = qbase + w * 16 + m;
            float tqv = tqs[w * 16 + m];
            float s0 = 0.25f + 0.25f * (c0[r] - tqv * tks[nl]);
            float s1 = 0.25f + 0.25f * (c1[r] - tqv * tks[16 + nl]);
            if (kpos0 + nl      > qg) s0 = -3.0e38f;
            if (kpos0 + 16 + nl > qg) s1 = -3.0e38f;
            float bm = fmaxf(s0, s1);
            #pragma unroll
            for (int off = 8; off; off >>= 1) bm = fmaxf(bm, __shfl_xor(bm, off, 32));
            float mold  = mrow[w][m];
            float mnew  = fmaxf(mold, bm);
            float alpha = __expf(mold - mnew);
            float p0 = __expf(s0 - mnew), p1 = __expf(s1 - mnew);
            float rs = p0 + p1;
            float rt = p0 * tvs[nl] + p1 * tvs[16 + nl];
            #pragma unroll
            for (int off = 8; off; off >>= 1) {
                rs += __shfl_xor(rs, off, 32);
                rt += __shfl_xor(rt, off, 32);
            }
            #pragma unroll
            for (int dt = 0; dt < 4; dt++) oac[dt][r] *= alpha;
            if (nl == 0) {
                mrow[w][m] = mnew;
                lrow[w][m] = lrow[w][m] * alpha + rs;
                trow[w][m] = trow[w][m] * alpha + rt;
            }
            Pst[w][m][nl]      = (_Float16)p0;
            Pst[w][m][16 + nl] = (_Float16)p1;
        }
        // O += P V
        v16h pa = loadA(&Pst[w][nl][0], 0, hf);
        #pragma unroll
        for (int dt = 0; dt < 4; dt++)
            oac[dt] = wmma_f16(pa, loadB(&Vt[dt * 16 + nl][0], 0, hf), oac[dt]);
    }

    // epilogue: normalize, Lorentz-project, drop time, store head space
    #pragma unroll
    for (int r = 0; r < 8; r++) {
        int m = r + 8 * hf;
        int qg = qbase + w * 16 + m;
        float linv = 1.0f / lrow[w][m];
        float tco  = trow[w][m] * linv;
        float ov[4]; float ss = 0.0f;
        #pragma unroll
        for (int dt = 0; dt < 4; dt++) { ov[dt] = oac[dt][r] * linv; ss += ov[dt] * ov[dt]; }
        #pragma unroll
        for (int off = 8; off; off >>= 1) ss += __shfl_xor(ss, off, 32);
        float dninv = rsqrtf(fmaxf(tco * tco - ss, 1e-6f));
        _Float16* op = Oout + (rowbase + qg) * D_MODEL + head * HD;
        #pragma unroll
        for (int dt = 0; dt < 4; dt++) op[dt * 16 + nl] = (_Float16)(ov[dt] * dninv);
    }
}

// ---------------------------------------------------------------------------
// x1 = proj(x + lift(g)/SCALE); also emit lrmsnorm(x1, ln2_w) f16 for the MLP
// ---------------------------------------------------------------------------
__global__ __launch_bounds__(256)
void resnet_norm(const float* __restrict__ G, const float* __restrict__ x,
                 const float* __restrict__ w2n, float* __restrict__ X1,
                 _Float16* __restrict__ Xn2)
{
    __shared__ float red[8];
    int row = blockIdx.x;
    const float* gr = G + (size_t)row * D_MODEL;   // cols 0..1022 valid
    const float* xr = x + (size_t)row * D_MODEL;
    float ssg = 0.0f;
    for (int j = threadIdx.x; j < D_MODEL - 1; j += 256) { float g = gr[j]; ssg += g * g; }
    ssg = block_sum(ssg, red);
    float t = sqrtf(ssg + 1.0f);
    float ssz = 0.0f;
    for (int j = threadIdx.x; j < D_MODEL - 1; j += 256) {
        float z = xr[1 + j] + gr[j] * INV_SCALE; ssz += z * z;
    }
    ssz = block_sum(ssz, red);
    float z0 = xr[0] + t * INV_SCALE;
    float dninv = rsqrtf(fmaxf(z0 * z0 - ssz, 1e-6f));
    float rmsinv = rsqrtf(ssz * dninv * dninv / (float)(D_MODEL - 1) + 1e-6f);
    float* o1 = X1 + (size_t)row * D_MODEL;
    _Float16* o2 = Xn2 + (size_t)row * D_MODEL;
    if (threadIdx.x == 0) o1[0] = z0 * dninv;
    for (int j = threadIdx.x; j < D_MODEL - 1; j += 256) {
        float z = xr[1 + j] + gr[j] * INV_SCALE;
        float s = z * dninv;
        o1[1 + j] = s;
        o2[j] = (_Float16)(s * rmsinv * w2n[j]);
    }
    if (threadIdx.x == 255) o2[D_MODEL - 1] = (_Float16)0.0f;
}

// ---------------------------------------------------------------------------
// out = proj(x1 + lift(y)/SCALE)
// ---------------------------------------------------------------------------
__global__ __launch_bounds__(256)
void final_out(const float* __restrict__ Y, const float* __restrict__ X1,
               float* __restrict__ out)
{
    __shared__ float red[8];
    int row = blockIdx.x;
    const float* yr = Y + (size_t)row * D_MODEL;
    const float* xr = X1 + (size_t)row * D_MODEL;
    float ssy = 0.0f;
    for (int j = threadIdx.x; j < D_MODEL - 1; j += 256) { float y = yr[j]; ssy += y * y; }
    ssy = block_sum(ssy, red);
    float t = sqrtf(ssy + 1.0f);
    float ssz = 0.0f;
    for (int j = threadIdx.x; j < D_MODEL - 1; j += 256) {
        float z = xr[1 + j] + yr[j] * INV_SCALE; ssz += z * z;
    }
    ssz = block_sum(ssz, red);
    float z0 = xr[0] + t * INV_SCALE;
    float dninv = rsqrtf(fmaxf(z0 * z0 - ssz, 1e-6f));
    float* orow = out + (size_t)row * D_MODEL;
    if (threadIdx.x == 0) orow[0] = z0 * dninv;
    for (int j = threadIdx.x; j < D_MODEL - 1; j += 256) {
        float z = xr[1 + j] + yr[j] * INV_SCALE;
        orow[1 + j] = z * dninv;
    }
}

// ---------------------------------------------------------------------------
extern "C" void kernel_launch(void* const* d_in, const int* in_sizes, int n_in,
                              void* d_out, int out_size, void* d_ws, size_t ws_size,
                              hipStream_t stream) {
    (void)in_sizes; (void)n_in; (void)out_size; (void)ws_size;
    const float* x    = (const float*)d_in[0];
    const float* ln1w = (const float*)d_in[1];
    const float* ln2w = (const float*)d_in[2];
    const float* wq   = (const float*)d_in[3];
    const float* wk   = (const float*)d_in[4];
    const float* wv   = (const float*)d_in[5];
    const float* wt   = (const float*)d_in[6];
    const float* w1   = (const float*)d_in[7];
    const float* b1   = (const float*)d_in[8];
    const float* w2   = (const float*)d_in[9];
    const float* b2   = (const float*)d_in[10];
    float* out = (float*)d_out;

    char* p = (char*)d_ws;
    auto take = [&](size_t bytes) -> char* {
        char* r = p; p += (bytes + 255) & ~(size_t)255; return r;
    };
    _Float16* WqT  = (_Float16*)take((size_t)1024 * 1024 * 2);
    _Float16* WkT  = (_Float16*)take((size_t)1024 * 1024 * 2);
    _Float16* WvT  = (_Float16*)take((size_t)1024 * 1024 * 2);
    _Float16* WtT  = (_Float16*)take((size_t)1024 * 1024 * 2);
    _Float16* W1T  = (_Float16*)take((size_t)4096 * 1024 * 2);
    _Float16* W2T  = (_Float16*)take((size_t)1024 * 4096 * 2);
    _Float16* Xn1  = (_Float16*)take((size_t)MROWS * 1024 * 2);
    _Float16* Qb   = (_Float16*)take((size_t)MROWS * 1024 * 2);
    _Float16* Kb   = (_Float16*)take((size_t)MROWS * 1024 * 2);
    _Float16* Vb   = (_Float16*)take((size_t)MROWS * 1024 * 2);
    float*    TQ   = (float*)take((size_t)MROWS * NHEAD * 4);
    float*    TK   = (float*)take((size_t)MROWS * NHEAD * 4);
    float*    TV   = (float*)take((size_t)MROWS * NHEAD * 4);
    _Float16* AttO = (_Float16*)take((size_t)MROWS * 1024 * 2);
    float*    Gbuf = (float*)take((size_t)MROWS * 1024 * 4);
    float*    X1b  = (float*)take((size_t)MROWS * 1024 * 4);
    _Float16* Xn2  = (_Float16*)take((size_t)MROWS * 1024 * 2);
    _Float16* Hb   = (_Float16*)take((size_t)MROWS * 4096 * 2);
    float*    Yb   = (float*)take((size_t)MROWS * 1024 * 4);

    // 1) pack weights (transposed to [N][Kpad], f16, zero-padded)
    pack_wT<<<2048, 256, 0, stream>>>(wq, WqT, 1023, 1024, 1024, 1024);
    pack_wT<<<2048, 256, 0, stream>>>(wk, WkT, 1023, 1024, 1024, 1024);
    pack_wT<<<2048, 256, 0, stream>>>(wv, WvT, 1023, 1024, 1024, 1024);
    pack_wT<<<2048, 256, 0, stream>>>(wt, WtT, 1024, 1023, 1024, 1024);
    pack_wT<<<4096, 256, 0, stream>>>(w1, W1T, 1023, 4096, 4096, 1024);
    pack_wT<<<4096, 256, 0, stream>>>(w2, W2T, 4096, 1023, 1024, 4096);

    // 2) lrmsnorm(x, ln1) -> f16 activations
    lrmsnorm_pack<<<MROWS, 256, 0, stream>>>(x, ln1w, Xn1);

    // 3) QKV projections (WMMA)
    gemm_wmma<<<dim3(8, 32), 256, 0, stream>>>(Xn1, WqT, Qb, nullptr, MROWS, 1024, 1024, 0, 1);
    gemm_wmma<<<dim3(8, 32), 256, 0, stream>>>(Xn1, WkT, Kb, nullptr, MROWS, 1024, 1024, 0, 1);
    gemm_wmma<<<dim3(8, 32), 256, 0, stream>>>(Xn1, WvT, Vb, nullptr, MROWS, 1024, 1024, 0, 1);

    // 4) per-head Lorentz time coordinates
    lift_t<<<(MROWS * NHEAD) / 256, 256, 0, stream>>>(Qb, TQ);
    lift_t<<<(MROWS * NHEAD) / 256, 256, 0, stream>>>(Kb, TK);
    lift_t<<<(MROWS * NHEAD) / 256, 256, 0, stream>>>(Vb, TV);

    // 5) causal Lorentz flash attention (WMMA) + proj
    attn_wmma<<<dim3(SS / 128, NHEAD, BB), 256, 0, stream>>>(Qb, Kb, Vb, TQ, TK, TV, AttO);

    // 6) output projection
    gemm_wmma<<<dim3(8, 32), 256, 0, stream>>>(AttO, WtT, Gbuf, nullptr, MROWS, 1024, 1024, 0, 0);

    // 7) residual+proj, then lrmsnorm(ln2) packed for MLP
    resnet_norm<<<MROWS, 256, 0, stream>>>(Gbuf, x, ln2w, X1b, Xn2);

    // 8) MLP up (gelu fused) and down (bias fused)
    gemm_wmma<<<dim3(32, 32), 256, 0, stream>>>(Xn2, W1T, Hb, b1, MROWS, 4096, 1024, 4096, 2);
    gemm_wmma<<<dim3(8, 32), 256, 0, stream>>>(Hb, W2T, Yb, b2, MROWS, 1024, 4096, 1023, 3);

    // 9) final residual + proj
    final_out<<<MROWS, 256, 0, stream>>>(Yb, X1b, out);
}
